// SplitHeadSamVisionSdpaAttention_10677288698167
// MI455X (gfx1250) — compile-verified, matrix operand order checked by reading, not run on hardware
//
#include <hip/hip_runtime.h>
#include <hip/hip_bf16.h>

// ---------------------------------------------------------------------------
// SAM vision attention on gfx1250 (MI455X).
//   * all GEMMs via v_wmma_f32_16x16x32_bf16 (f32 accumulate)
//   * attn (255 MB) writeback via global_store_async_from_lds_b128, overlapped
//     with the PV WMMA loop, fenced by s_wait_asynccnt
// ---------------------------------------------------------------------------

typedef __bf16 bf16;
typedef bf16  v16bf __attribute__((ext_vector_type(16)));
typedef float v8f   __attribute__((ext_vector_type(8)));

#define HW_TOK   2304   // 48*48
#define EMBED_C  768
#define NHEAD    12
#define HDIM     64
#define SIDE     48
#define SROW     2308   // padded LDS row stride (floats): 4*m mod 64 distinct -> no bank conflicts

__device__ __forceinline__ v8f wmma_bf16(v16bf a, v16bf b, v8f c) {
    // (neg_a, A, neg_b, B, c_mod, C, reuse_a, reuse_b)
    return __builtin_amdgcn_wmma_f32_16x16x32_bf16(false, a, false, b, (short)0, c,
                                                   false, false);
}

// ---------------------------------------------------------------------------
// Kernel 1: QKV projection.  grid = (144 t-tiles, 12 heads, 3 matrices), 128 thr.
// y[t,o] = sum_c x[t,c] * w[o,c] + b[o]   (NT GEMM, both operands K-contiguous)
// q -> f32 [h][t][d], k -> bf16 [h][t][d], v -> bf16 transposed [h][d][t]
// ---------------------------------------------------------------------------
__global__ __launch_bounds__(128)
void qkv_kernel(const float* __restrict__ x,
                const float* __restrict__ wq, const float* __restrict__ bq,
                const float* __restrict__ wk, const float* __restrict__ bk,
                const float* __restrict__ wv, const float* __restrict__ bv,
                float* __restrict__ qf, bf16* __restrict__ kb,
                bf16* __restrict__ vT)
{
    const int lane  = threadIdx.x & 31;
    const int wave  = threadIdx.x >> 5;     // 0..3 : 16-wide d-subtile
    const int mat   = blockIdx.z;           // 0=q 1=k 2=v
    const int head  = blockIdx.y;

    const float* w  = (mat == 0) ? wq : (mat == 1) ? wk : wv;
    const float* bs = (mat == 0) ? bq : (mat == 1) ? bk : bv;

    const int row   = lane & 15;
    const int khalf = (lane >> 4) * 16;
    const int t0    = blockIdx.x * 16;
    const int o0    = head * HDIM + wave * 16;

    v8f acc = {};
    for (int kk = 0; kk < EMBED_C; kk += 32) {
        const float* ap = x + (size_t)(t0 + row) * EMBED_C + kk + khalf;
        const float* bp = w + (size_t)(o0 + row) * EMBED_C + kk + khalf;
        v16bf a, b;
#pragma unroll
        for (int i = 0; i < 16; ++i) { a[i] = (bf16)ap[i]; b[i] = (bf16)bp[i]; }
        acc = wmma_bf16(a, b, acc);
    }

    const int n   = lane & 15;
    const int hi8 = (lane >> 4) * 8;
    const float bias = bs[o0 + n];
#pragma unroll
    for (int r = 0; r < 8; ++r) acc[r] += bias;

    // uniform branch OUTSIDE the unrolled store loops -> straight-line stores
    if (mat == 0) {
        float* dst = qf + ((size_t)head * HW_TOK + t0 + hi8) * HDIM + wave * 16 + n;
#pragma unroll
        for (int r = 0; r < 8; ++r) dst[(size_t)r * HDIM] = acc[r];
    } else if (mat == 1) {
        bf16* dst = kb + ((size_t)head * HW_TOK + t0 + hi8) * HDIM + wave * 16 + n;
#pragma unroll
        for (int r = 0; r < 8; ++r) dst[(size_t)r * HDIM] = (bf16)acc[r];
    } else {
        // vT[h][d][t]: fixed d per lane, 8 consecutive t -> 16B contiguous store
        bf16* dst = vT + ((size_t)head * HDIM + wave * 16 + n) * HW_TOK + t0 + hi8;
#pragma unroll
        for (int r = 0; r < 8; ++r) dst[r] = (bf16)acc[r];
    }
}

// ---------------------------------------------------------------------------
// Kernel 2: decomposed rel-pos dot products (~0.34 GF, plain VALU).
// rel_h[h][q][ky] = sum_c q[h][q][c] * rel_pos_h[qy-ky+47][c]
// ---------------------------------------------------------------------------
__global__ __launch_bounds__(64)
void relpos_kernel(const float* __restrict__ qf,
                   const float* __restrict__ rph, const float* __restrict__ rpw,
                   float* __restrict__ rel_h, float* __restrict__ rel_w)
{
    const int q  = blockIdx.x;
    const int bh = blockIdx.y;
    const int k  = threadIdx.x;
    if (k >= SIDE) return;
    const int qy = q / SIDE, qx = q % SIDE;
    const float* qv = qf  + ((size_t)bh * HW_TOK + q) * HDIM;
    const float* rh = rph + (size_t)(qy - k + SIDE - 1) * HDIM;
    const float* rw = rpw + (size_t)(qx - k + SIDE - 1) * HDIM;
    float sh = 0.f, sw = 0.f;
#pragma unroll 8
    for (int c = 0; c < HDIM; ++c) {
        const float qc = qv[c];
        sh = fmaf(qc, rh[c], sh);
        sw = fmaf(qc, rw[c], sw);
    }
    rel_h[((size_t)bh * HW_TOK + q) * SIDE + k] = sh;
    rel_w[((size_t)bh * HW_TOK + q) * SIDE + k] = sw;
}

// ---------------------------------------------------------------------------
// Kernel 3: attention core.  grid = (144 q-tiles, 12 heads), 128 threads.
// LDS score strip 16 x SROW f32 (~147.7 KB, fits CDNA5 320KB WGP LDS).
// scores(QK^T)+bias -> softmax -> async LDS->HBM attn writeback (overlapped)
// -> PV WMMA -> s_wait_asynccnt.
// ---------------------------------------------------------------------------
__global__ __launch_bounds__(128)
void attn_kernel(const float* __restrict__ qf, const bf16* __restrict__ kb,
                 const bf16* __restrict__ vT,
                 const float* __restrict__ rel_h, const float* __restrict__ rel_w,
                 float* __restrict__ attn_out,     // (12, 2304, 2304)
                 float* __restrict__ stage)        // (2304, 768) head-major channels
{
    extern __shared__ float s[];                   // [16][SROW]
    __shared__ float relh_s[16 * SIDE];            // bias tables for this q-tile
    __shared__ float relw_s[16 * SIDE];
    __shared__ float red[16 * 8];
    __shared__ float rowmax[16], rowinv[16];

    const int lane  = threadIdx.x & 31;
    const int wave  = threadIdx.x >> 5;
    const int qt    = blockIdx.x;
    const int head  = blockIdx.y;

    const int row   = lane & 15;
    const int hi    = lane >> 4;
    const int khalf = hi * 16;
    const int q0    = qt * 16;

    // preload per-tile rel bias tables (2 x 768 floats) into LDS
    for (int idx = threadIdx.x; idx < 16 * SIDE; idx += 128) {
        const size_t qrow = (size_t)head * HW_TOK + q0 + (idx / SIDE);
        relh_s[idx] = rel_h[qrow * SIDE + (idx % SIDE)];
        relw_s[idx] = rel_w[qrow * SIDE + (idx % SIDE)];
    }

    // Scaled Q tile as WMMA A operand (K=64 split into two K=32 halves).
    v16bf a0, a1;
    {
        const float* qp = qf + ((size_t)head * HW_TOK + q0 + row) * HDIM;
#pragma unroll
        for (int i = 0; i < 16; ++i) {
            a0[i] = (bf16)(qp[khalf + i]      * 0.125f);   // SCALE = 64^-0.5
            a1[i] = (bf16)(qp[32 + khalf + i] * 0.125f);
        }
    }
    __syncthreads();

    // ---- scores + bias: 4 waves sweep 144 key tiles ----
    for (int kt = wave; kt < HW_TOK / 16; kt += 4) {
        const int k0  = kt * 16;
        const int ky  = k0 / SIDE;          // constant per key tile (16 | 48)
        const int kx0 = k0 % SIDE;
        v16bf b0, b1;
        const bf16* kp = kb + ((size_t)head * HW_TOK + k0 + row) * HDIM;
#pragma unroll
        for (int i = 0; i < 16; ++i) { b0[i] = kp[khalf + i]; b1[i] = kp[32 + khalf + i]; }
        v8f c = {};
        c = wmma_bf16(a0, b0, c);
        c = wmma_bf16(a1, b1, c);

        const int n  = lane & 15;
        const int kx = kx0 + n;
#pragma unroll
        for (int r = 0; r < 8; ++r) {
            const int m = r + 8 * hi;
            s[m * SROW + k0 + n] = c[r] + relh_s[m * SIDE + ky] + relw_s[m * SIDE + kx];
        }
    }
    __syncthreads();

    // ---- softmax: 8 threads per row ----
    {
        const int r = threadIdx.x >> 3, seg = threadIdx.x & 7;
        float mx = -3.0e38f;
        for (int j = seg; j < HW_TOK; j += 8) mx = fmaxf(mx, s[r * SROW + j]);
        red[r * 8 + seg] = mx;
    }
    __syncthreads();
    if (threadIdx.x < 16) {
        float mx = red[threadIdx.x * 8];
#pragma unroll
        for (int j = 1; j < 8; ++j) mx = fmaxf(mx, red[threadIdx.x * 8 + j]);
        rowmax[threadIdx.x] = mx;
    }
    __syncthreads();
    {
        const int r = threadIdx.x >> 3, seg = threadIdx.x & 7;
        const float mx = rowmax[r];
        float sum = 0.f;
        for (int j = seg; j < HW_TOK; j += 8) {
            const float e = __expf(s[r * SROW + j] - mx);
            s[r * SROW + j] = e;
            sum += e;
        }
        red[r * 8 + seg] = sum;
    }
    __syncthreads();
    if (threadIdx.x < 16) {
        float sm = 0.f;
#pragma unroll
        for (int j = 0; j < 8; ++j) sm += red[threadIdx.x * 8 + j];
        rowinv[threadIdx.x] = 1.0f / sm;
    }
    __syncthreads();

    // ---- normalize probabilities in LDS ----
    for (int m = 0; m < 16; ++m) {
        const float inv = rowinv[m];
        for (int j = threadIdx.x; j < HW_TOK; j += 128) s[m * SROW + j] *= inv;
    }
    __syncthreads();

    // ---- async attn writeback: LDS -> HBM, 72 x b128 per lane, uniform loop ----
    // global rows are contiguous (16 x 2304 f32 = 9216 b128 = 72 iters x 128 lanes);
    // overlaps with the PV WMMA loop below, fenced by s_wait_asynccnt at the end.
    {
        const uint64_t gbase =
            (uint64_t)(uintptr_t)(attn_out + ((size_t)head * HW_TOK + q0) * (size_t)HW_TOK);
        const unsigned lbase = (unsigned)(uintptr_t)s;   // LDS byte offset (low 32 bits)
        for (int it = 0; it < 72; ++it) {
            const int bidx = threadIdx.x + it * 128;     // b128 index
            const int m    = bidx / 576;                 // 576 b128 per row
            const int j16  = bidx - m * 576;
            const unsigned lds_addr = lbase + (unsigned)(m * (SROW * 4) + j16 * 16);
            const uint64_t ga = gbase + (uint64_t)bidx * 16u;
            asm volatile("global_store_async_from_lds_b128 %0, %1, off"
                         :: "v"(ga), "v"(lds_addr) : "memory");
        }
    }

    // ---- PV: out(16x64) = probs(16x2304) x V(2304x64); wave owns 16 dims ----
    v8f o = {};
    for (int kk = 0; kk < HW_TOK; kk += 32) {
        v16bf a, b;
        const float* sp = s + row * SROW + kk + khalf;
#pragma unroll
        for (int i = 0; i < 16; ++i) a[i] = (bf16)sp[i];
        const bf16* vp = vT + ((size_t)head * HDIM + wave * 16 + row) * HW_TOK + kk + khalf;
#pragma unroll
        for (int i = 0; i < 16; ++i) b[i] = vp[i];
        o = wmma_bf16(a, b, o);
    }
    {
        const int n = lane & 15;
#pragma unroll
        for (int r = 0; r < 8; ++r) {
            const int m = r + 8 * hi;
            stage[(size_t)(q0 + m) * EMBED_C + head * HDIM + wave * 16 + n] = o[r];
        }
    }

    // drain the async LDS->memory DMA before the wave retires
    asm volatile("s_wait_asynccnt 0x0" ::: "memory");
}

// ---------------------------------------------------------------------------
// Kernel 4: output projection, same WMMA NT-GEMM as kernel 1.
// grid = (144 t-tiles, 12 o64-tiles), 128 threads.
// ---------------------------------------------------------------------------
__global__ __launch_bounds__(128)
void proj_kernel(const float* __restrict__ stage,
                 const float* __restrict__ wp, const float* __restrict__ bp,
                 float* __restrict__ out)
{
    const int lane  = threadIdx.x & 31;
    const int wave  = threadIdx.x >> 5;
    const int row   = lane & 15;
    const int khalf = (lane >> 4) * 16;
    const int t0    = blockIdx.x * 16;
    const int o0    = blockIdx.y * 64 + wave * 16;

    v8f acc = {};
    for (int kk = 0; kk < EMBED_C; kk += 32) {
        const float* ap  = stage + (size_t)(t0 + row) * EMBED_C + kk + khalf;
        const float* wpp = wp    + (size_t)(o0 + row) * EMBED_C + kk + khalf;
        v16bf a, b;
#pragma unroll
        for (int i = 0; i < 16; ++i) { a[i] = (bf16)ap[i]; b[i] = (bf16)wpp[i]; }
        acc = wmma_bf16(a, b, acc);
    }
    const int n = lane & 15;
    const float bias = bp[o0 + n];
#pragma unroll
    for (int r = 0; r < 8; ++r) {
        const int m = r + 8 * (lane >> 4);
        out[(size_t)(t0 + m) * EMBED_C + o0 + n] = acc[r] + bias;
    }
}

// ---------------------------------------------------------------------------
extern "C" void kernel_launch(void* const* d_in, const int* in_sizes, int n_in,
                              void* d_out, int out_size, void* d_ws, size_t ws_size,
                              hipStream_t stream)
{
    const float* x    = (const float*)d_in[0];   // (1,48,48,768)
    const float* wq   = (const float*)d_in[1];
    const float* bq   = (const float*)d_in[2];
    const float* wk   = (const float*)d_in[3];
    const float* bk   = (const float*)d_in[4];
    const float* wv   = (const float*)d_in[5];
    const float* bv   = (const float*)d_in[6];
    const float* wpj  = (const float*)d_in[7];
    const float* bpj  = (const float*)d_in[8];
    const float* rph  = (const float*)d_in[9];   // (95,64)
    const float* rpw  = (const float*)d_in[10];  // (95,64)

    float* out_final = (float*)d_out;                              // 2304*768
    float* attn_out  = out_final + (size_t)HW_TOK * EMBED_C;       // 12*2304*2304

    // ---- workspace carve (~32 MB) ----
    char* ws = (char*)d_ws;
    float* qf    = (float*)ws; ws += (size_t)NHEAD * HW_TOK * HDIM * sizeof(float);
    bf16*  kb    = (bf16*) ws; ws += (size_t)NHEAD * HW_TOK * HDIM * sizeof(bf16);
    bf16*  vT    = (bf16*) ws; ws += (size_t)NHEAD * HDIM * HW_TOK * sizeof(bf16);
    float* rel_h = (float*)ws; ws += (size_t)NHEAD * HW_TOK * SIDE * sizeof(float);
    float* rel_w = (float*)ws; ws += (size_t)NHEAD * HW_TOK * SIDE * sizeof(float);
    float* stage = (float*)ws; // 2304*768 f32

    // 1) QKV projections (WMMA bf16)
    qkv_kernel<<<dim3(HW_TOK / 16, NHEAD, 3), 128, 0, stream>>>(
        x, wq, bq, wk, bk, wv, bv, qf, kb, vT);

    // 2) rel-pos bias tables
    relpos_kernel<<<dim3(HW_TOK, NHEAD), 64, 0, stream>>>(qf, rph, rpw, rel_h, rel_w);

    // 3) attention core (async attn writeback overlapped with PV WMMA)
    const size_t smem = (size_t)16 * SROW * sizeof(float);
    attn_kernel<<<dim3(HW_TOK / 16, NHEAD), 128, smem, stream>>>(
        qf, kb, vT, rel_h, rel_w, attn_out, stage);

    // 4) output projection
    proj_kernel<<<dim3(HW_TOK / 16, EMBED_C / 64), 128, 0, stream>>>(
        stage, wpj, bpj, out_final);
}
